// RnnModelPT_82901458748038
// MI455X (gfx1250) — matrix operand order repaired and checked by reading, not compiled, not run on hardware
//
#include <hip/hip_runtime.h>

typedef __bf16 bf16;
typedef __attribute__((ext_vector_type(16))) __bf16 v16bf;
typedef __attribute__((ext_vector_type(8)))  __bf16 v8bf;
typedef __attribute__((ext_vector_type(8)))  float   v8f;

constexpr int BB = 256;     // batch
constexpr int SS = 2048;    // sequence
constexpr int DD = 64;      // input dim
constexpr int HH = 256;     // hidden
constexpr int LDH = HH + 8; // padded LDS row stride (breaks 512B bank stride)
constexpr int CH = 16;      // recurrence steps per pipeline chunk
constexpr int NC = SS / CH; // 128 chunks
constexpr int NBLK = 16;    // batch-tile blocks per pipeline stage
constexpr int TFRAG = 2 * 8 * 2 * 32 * 8 / 2; // 4096 bf16 elems per (s,j) pre1 tile

// D = A(16x32 bf16) * B(32x16 bf16) + C(16x16 f32) -- probe-confirmed builtin
__device__ __forceinline__ v8f wmma_b16(v16bf a, v16bf b, v8f c) {
  return __builtin_amdgcn_wmma_f32_16x16x32_bf16(false, a, false, b, (short)0, c,
                                                 false, false);
}

__device__ __forceinline__ v16bf a_pack(v8bf lo, v8bf hi) {
  return __builtin_shufflevector(lo, hi, 0, 1, 2, 3, 4, 5, 6, 7,
                                         8, 9, 10, 11, 12, 13, 14, 15);
}

// gfx1250 native v_tanh_f32 (TRANS32); libm fallback elsewhere.
__device__ __forceinline__ float fast_tanh(float x) {
#if __has_builtin(__builtin_amdgcn_tanhf)
  return __builtin_amdgcn_tanhf(x);
#else
  return tanhf(x);
#endif
}

// A-fragment (16x32, 16-bit layout) of a weight tile, from f32 row-major W.
// prow must already point at this lane's matrix row (n = ntile*16 + (lane&15)).
// lanes 0-15: K = k0..k0+15; lanes 16-31: K = k0+16..k0+31.
__device__ __forceinline__ v16bf a_frag_f32(const float* __restrict__ prow,
                                            int k0, int hi) {
  const float* p = prow + k0 + hi * 16;
  v16bf r;
#pragma unroll
  for (int e = 0; e < 8; ++e) r[e] = (bf16)p[e];
#pragma unroll
  for (int e = 0; e < 8; ++e) r[e + 8] = (bf16)p[8 + e];
  return r;
}

// B-fragment (32x16): lane = N column (batch row m), elements = 16 contiguous K.
// prow points at this lane's h row; 32B contiguous -> two 16B loads.
__device__ __forceinline__ v16bf b_frag_b16(const bf16* __restrict__ prow,
                                            int k0, int hi) {
  const bf16* p = prow + k0 + hi * 16;
  v8bf lo = *(const v8bf*)p;
  v8bf h2 = *(const v8bf*)(p + 8);
  return a_pack(lo, h2);
}

// B-fragment from f32 data (x input): 16 contiguous floats -> bf16.
__device__ __forceinline__ v16bf b_frag_f32(const float* __restrict__ p) {
  v16bf r;
#pragma unroll
  for (int e = 0; e < 16; ++e) r[e] = (bf16)p[e];
  return r;
}

// ---- cross-workgroup chunk flags (producer/consumer pipeline) -------------
__device__ __forceinline__ void flag_wait(const int* f) {
  if (threadIdx.x == 0) {
    while (__hip_atomic_load(f, __ATOMIC_RELAXED, __HIP_MEMORY_SCOPE_AGENT) == 0)
      __builtin_amdgcn_s_sleep(2);
  }
  __syncthreads();
  __builtin_amdgcn_fence(__ATOMIC_ACQUIRE, "agent");
}

__device__ __forceinline__ void flag_set(int* f) {
  __threadfence();  // each thread drains its own stores to device scope
  __syncthreads();
  if (threadIdx.x == 0)
    __hip_atomic_store(f, 1, __ATOMIC_RELEASE, __HIP_MEMORY_SCOPE_AGENT);
}

__global__ void k_zero_flags(int* __restrict__ f, int n) {
  const int i = blockIdx.x * blockDim.x + threadIdx.x;
  if (i < n) f[i] = 0;
}

// ---------------------------------------------------------------------------
// Fused pipeline, 48 blocks x 256 threads. GEMMs are formulated transposed,
// D^T = W x h^T, so that the WMMA D layout (lane = batch row, vgprs = 8
// consecutive n) lets every lane write h_new / pre1 as single 16B vectors.
//   role 0: layer-0 recurrence + fused x-projection; streams ys0, chunk flags.
//   role 1: pre1 = ys0@Wih1^T + b1 in fragment-major layout, behind role 0.
//   role 2: layer-1 recurrence + sigmoid head.
// ---------------------------------------------------------------------------
__global__ __launch_bounds__(256) void k_pipeline(
    const float* __restrict__ x,
    const float* __restrict__ Wih0, const float* __restrict__ Whh0,
    const float* __restrict__ bih0, const float* __restrict__ bhh0,
    const float* __restrict__ Wih1, const float* __restrict__ Whh1,
    const float* __restrict__ bih1, const float* __restrict__ bhh1,
    const float* __restrict__ Wout, const float* __restrict__ bout,
    bf16* __restrict__ ys0, bf16* __restrict__ pre1,
    int* __restrict__ flags0, int* __restrict__ flags1,
    float* __restrict__ out) {
  __shared__ __attribute__((aligned(16))) bf16 hbuf[2][16][LDH];
  const int tid = threadIdx.x;
  const int lane = tid & 31;
  const int wave = tid >> 5;
  const int m = lane & 15;          // batch row within tile (B-frag column)
  const int hi = (lane >> 4) & 1;
  const int role = blockIdx.x >> 4;
  const int j = blockIdx.x & 15;
  const int b0 = j * 16;
  const int n0 = (wave * 2 + 0) * 16 + m;  // A-frag row for n-tile 0
  const int n1 = (wave * 2 + 1) * 16 + m;  // A-frag row for n-tile 1
  // this lane's 8 output columns per n-tile start at:
  const int c0 = (wave * 2 + 0) * 16 + hi * 8;
  const int c1 = (wave * 2 + 1) * 16 + hi * 8;
  // fragment-major offsets for the pre1 handoff (one 16B vector per nt)
  const size_t f0 = (size_t)((wave * 2 + 0) * 32 + lane) * 8;
  const size_t f1 = (size_t)((wave * 2 + 1) * 32 + lane) * 8;

  if (role == 0) {
    // ---------------- stage 0: layer-0 recurrence (producer) ---------------
    for (int i = tid; i < 2 * 16 * LDH; i += 256) ((bf16*)hbuf)[i] = (bf16)0.f;
    v16bf Wr[2][8], Wi[2][2];
#pragma unroll
    for (int kt = 0; kt < 8; ++kt) {
      Wr[0][kt] = a_frag_f32(Whh0 + (size_t)n0 * HH, kt * 32, hi);
      Wr[1][kt] = a_frag_f32(Whh0 + (size_t)n1 * HH, kt * 32, hi);
    }
#pragma unroll
    for (int kt = 0; kt < 2; ++kt) {
      Wi[0][kt] = a_frag_f32(Wih0 + (size_t)n0 * DD, kt * 32, hi);
      Wi[1][kt] = a_frag_f32(Wih0 + (size_t)n1 * DD, kt * 32, hi);
    }
    float bias[2][8];
#pragma unroll
    for (int r = 0; r < 8; ++r) {
      bias[0][r] = bih0[c0 + r] + bhh0[c0 + r];
      bias[1][r] = bih0[c1 + r] + bhh0[c1 + r];
    }
    const float* xrow = x + (size_t)(b0 + m) * SS * DD;  // x is [B][S][D]
    __syncthreads();

    v16bf xb[2] = {b_frag_f32(xrow + 0 + hi * 16),
                   b_frag_f32(xrow + 32 + hi * 16)};
    for (int c = 0; c < NC; ++c) {
      for (int t = 0; t < CH; ++t) {
        const int s = c * CH + t;
        const int p = s & 1;
        const bool nx = (s + 1 < SS);
        v16bf xn[2];
        if (nx) {  // prefetch next step's x fragments (off-chain)
          xn[0] = b_frag_f32(xrow + (size_t)(s + 1) * DD + 0 + hi * 16);
          xn[1] = b_frag_f32(xrow + (size_t)(s + 1) * DD + 32 + hi * 16);
        }
        // four independent accumulation chains to hide WMMA RAW latency
        v8f a0a = {}, a0b = {}, a1a = {}, a1b = {};
#pragma unroll
        for (int kt = 0; kt < 2; ++kt) {  // fused input projection, K=64
          a0a = wmma_b16(Wi[0][kt], xb[kt], a0a);
          a1a = wmma_b16(Wi[1][kt], xb[kt], a1a);
        }
#pragma unroll
        for (int kt = 0; kt < 4; ++kt) {  // recurrent matmul, K lo half
          v16bf hB = b_frag_b16(&hbuf[p][m][0], kt * 32, hi);
          a0a = wmma_b16(Wr[0][kt], hB, a0a);
          a1a = wmma_b16(Wr[1][kt], hB, a1a);
        }
#pragma unroll
        for (int kt = 4; kt < 8; ++kt) {  // recurrent matmul, K hi half
          v16bf hB = b_frag_b16(&hbuf[p][m][0], kt * 32, hi);
          a0b = wmma_b16(Wr[0][kt], hB, a0b);
          a1b = wmma_b16(Wr[1][kt], hB, a1b);
        }
        const v8f acc0 = a0a + a0b, acc1 = a1a + a1b;
        v8bf h0v, h1v;
#pragma unroll
        for (int r = 0; r < 8; ++r) {
          h0v[r] = (bf16)fast_tanh(acc0[r] + bias[0][r]);
          h1v[r] = (bf16)fast_tanh(acc1[r] + bias[1][r]);
        }
        *(v8bf*)&hbuf[1 - p][m][c0] = h0v;  // one 16B LDS store per n-tile
        *(v8bf*)&hbuf[1 - p][m][c1] = h1v;
        __syncthreads();
        {  // coalesced stream-out of h_t (row-major, for stage-1 B-frags)
          const int mm = tid >> 4, cc = tid & 15;
          const bf16* src = &hbuf[1 - p][mm][cc * 16];
          bf16* dst = ys0 + ((size_t)s * BB + b0 + mm) * HH + cc * 16;
          *(v8bf*)dst = *(const v8bf*)src;
          *(v8bf*)(dst + 8) = *(const v8bf*)(src + 8);
        }
        if (nx) { xb[0] = xn[0]; xb[1] = xn[1]; }
      }
      flag_set(&flags0[j * NC + c]);
    }
  } else if (role == 1) {
    // ------------- stage 1: layer-1 input projection (throughput) ----------
    v16bf Wp[2][8];
#pragma unroll
    for (int kt = 0; kt < 8; ++kt) {
      Wp[0][kt] = a_frag_f32(Wih1 + (size_t)n0 * HH, kt * 32, hi);
      Wp[1][kt] = a_frag_f32(Wih1 + (size_t)n1 * HH, kt * 32, hi);
    }
    float bias[2][8];
#pragma unroll
    for (int r = 0; r < 8; ++r) {
      bias[0][r] = bih1[c0 + r] + bhh1[c0 + r];
      bias[1][r] = bih1[c1 + r] + bhh1[c1 + r];
    }
    for (int c = 0; c < NC; ++c) {
      flag_wait(&flags0[j * NC + c]);
      for (int t = 0; t < CH; ++t) {
        const int s = c * CH + t;
        const bf16* yrow = ys0 + ((size_t)s * BB + b0 + m) * HH;
        v8f a0a = {}, a0b = {}, a1a = {}, a1b = {};
#pragma unroll
        for (int kt = 0; kt < 4; ++kt) {
          v16bf yB = b_frag_b16(yrow, kt * 32, hi);
          a0a = wmma_b16(Wp[0][kt], yB, a0a);
          a1a = wmma_b16(Wp[1][kt], yB, a1a);
        }
#pragma unroll
        for (int kt = 4; kt < 8; ++kt) {
          v16bf yB = b_frag_b16(yrow, kt * 32, hi);
          a0b = wmma_b16(Wp[0][kt], yB, a0b);
          a1b = wmma_b16(Wp[1][kt], yB, a1b);
        }
        const v8f acc0 = a0a + a0b, acc1 = a1a + a1b;
        v8bf p0v, p1v;
#pragma unroll
        for (int r = 0; r < 8; ++r) {
          p0v[r] = (bf16)(acc0[r] + bias[0][r]);
          p1v[r] = (bf16)(acc1[r] + bias[1][r]);
        }
        bf16* pb = pre1 + ((size_t)s * NBLK + j) * TFRAG;
        *(v8bf*)(pb + f0) = p0v;  // fragment-major: one 16B store per n-tile
        *(v8bf*)(pb + f1) = p1v;
      }
      flag_set(&flags1[j * NC + c]);
    }
  } else {
    // ---------------- stage 2: layer-1 recurrence + head -------------------
    for (int i = tid; i < 2 * 16 * LDH; i += 256) ((bf16*)hbuf)[i] = (bf16)0.f;
    v16bf Wr[2][8];
#pragma unroll
    for (int kt = 0; kt < 8; ++kt) {
      Wr[0][kt] = a_frag_f32(Whh1 + (size_t)n0 * HH, kt * 32, hi);
      Wr[1][kt] = a_frag_f32(Whh1 + (size_t)n1 * HH, kt * 32, hi);
    }
    __syncthreads();

    flag_wait(&flags1[j * NC + 0]);
    v8bf pc0 = *(const v8bf*)(pre1 + ((size_t)0 * NBLK + j) * TFRAG + f0);
    v8bf pc1 = *(const v8bf*)(pre1 + ((size_t)0 * NBLK + j) * TFRAG + f1);
    for (int c = 0; c < NC; ++c) {
      const int cw = (c + 1 < NC) ? (c + 1) : (NC - 1);
      flag_wait(&flags1[j * NC + cw]);  // covers chunk c + prefetch into c+1
      for (int t = 0; t < CH; ++t) {
        const int s = c * CH + t;
        const int p = s & 1;
        const bool nx = (s + 1 < SS);
        v8bf pn0, pn1;
        if (nx) {  // single 16B prefetch per n-tile
          const bf16* pb = pre1 + ((size_t)(s + 1) * NBLK + j) * TFRAG;
          pn0 = *(const v8bf*)(pb + f0);
          pn1 = *(const v8bf*)(pb + f1);
        }
        v8f a0a = {}, a0b = {}, a1a = {}, a1b = {};
#pragma unroll
        for (int kt = 0; kt < 4; ++kt) {
          v16bf hB = b_frag_b16(&hbuf[p][m][0], kt * 32, hi);
          a0a = wmma_b16(Wr[0][kt], hB, a0a);
          a1a = wmma_b16(Wr[1][kt], hB, a1a);
        }
#pragma unroll
        for (int kt = 4; kt < 8; ++kt) {
          v16bf hB = b_frag_b16(&hbuf[p][m][0], kt * 32, hi);
          a0b = wmma_b16(Wr[0][kt], hB, a0b);
          a1b = wmma_b16(Wr[1][kt], hB, a1b);
        }
        const v8f acc0 = a0a + a0b, acc1 = a1a + a1b;
        v8bf h0v, h1v;
#pragma unroll
        for (int r = 0; r < 8; ++r) {
          h0v[r] = (bf16)fast_tanh(acc0[r] + (float)pc0[r]);
          h1v[r] = (bf16)fast_tanh(acc1[r] + (float)pc1[r]);
        }
        *(v8bf*)&hbuf[1 - p][m][c0] = h0v;
        *(v8bf*)&hbuf[1 - p][m][c1] = h1v;
        __syncthreads();
        if (nx) { pc0 = pn0; pc1 = pn1; }
      }
    }
    // out[b] = sigmoid(h_last . W_out + b_out); final h lives in hbuf[SS & 1].
    if (tid < 16) {
      const bf16* hrow = hbuf[SS & 1][tid];
      float acc = 0.f;
      for (int k = 0; k < HH; ++k) acc += (float)hrow[k] * Wout[k];
      out[b0 + tid] = 1.0f / (1.0f + __expf(-(acc + bout[0])));
    }
  }
}

// ---------------------------------------------------------------------------
// Launcher. Workspace: ys0 [S*B*H] bf16 (256 MB) + pre1 fragment-major
// [S][16][4096] bf16 (256 MB) + 2*16*128 int flags (16 KB, re-zeroed every
// launch so graph replays are self-contained).
// ---------------------------------------------------------------------------
extern "C" void kernel_launch(void* const* d_in, const int* in_sizes, int n_in,
                              void* d_out, int out_size, void* d_ws,
                              size_t ws_size, hipStream_t stream) {
  (void)in_sizes; (void)n_in; (void)out_size; (void)ws_size;
  const float* x    = (const float*)d_in[0];
  const float* Wih0 = (const float*)d_in[1];
  const float* Whh0 = (const float*)d_in[2];
  const float* bih0 = (const float*)d_in[3];
  const float* bhh0 = (const float*)d_in[4];
  const float* Wih1 = (const float*)d_in[5];
  const float* Whh1 = (const float*)d_in[6];
  const float* bih1 = (const float*)d_in[7];
  const float* bhh1 = (const float*)d_in[8];
  const float* Wout = (const float*)d_in[9];
  const float* bout = (const float*)d_in[10];
  float* out = (float*)d_out;

  bf16* ys0  = (bf16*)d_ws;
  bf16* pre1 = ys0 + (size_t)SS * BB * HH;
  int*  flags0 = (int*)(pre1 + (size_t)SS * NBLK * TFRAG);
  int*  flags1 = flags0 + NBLK * NC;
  const int nflags = 2 * NBLK * NC;

  k_zero_flags<<<(nflags + 255) / 256, 256, 0, stream>>>(flags0, nflags);
  k_pipeline<<<3 * NBLK, 256, 0, stream>>>(x, Wih0, Whh0, bih0, bhh0, Wih1,
                                           Whh1, bih1, bhh1, Wout, bout, ys0,
                                           pre1, flags0, flags1, out);
}
